// EarthAttention3D_64828236365993
// MI455X (gfx1250) — matrix operand order repaired and checked by reading, not compiled, not run on hardware
//
#include <hip/hip_runtime.h>

// EarthAttention3D (Pangu-Weather) fully fused, bf16 WMMA, gfx1250.
// B_=15, nW=64, N=144 tokens, C=192, 6 heads, hd=32. One block per (b,w).

#define DIM    192
#define NH     6
#define HD     32
#define NTOK   144
#define NTOKP  160          // tokens padded to multiple of 32 for K-chunks
#define NWT    64
#define NLON   15
#define NTHREADS 288        // 9 waves; each wave owns 16 of the 144 rows
#define NWAVES 9

typedef __attribute__((ext_vector_type(16))) __bf16 v16bf;
typedef __attribute__((ext_vector_type(8)))  float  v8f;

// ---- LDS layout (bytes) ------------------------------------------------
#define OFF_XS   0          // x tile bf16 [144][192]            55296
#define OFF_AO   55296      // attn output bf16 [144][192]       55296
#define OFF_PS   110592     // probabilities bf16 [144][160]     46080
#define OFF_VTS  156672     // v^T bf16 [32][160]                10240
#define OFF_QS   166912     // q bf16 [144][32]                   9216
#define OFF_KS   176128     // k bf16 [144][32]                   9216
#define OFF_WT   185344     // qkv W^T slices bf16 [3][32][192]  36864
#define OFF_PWT  110592     // proj_w^T bf16 [192][192] 73728 (aliases PS..KS, dead by then)
#define SMEM_BYTES 222208

// A fragment (16x32 bf16): lane m = m0 + (lane&15); khalf = 8*(lane>=16);
// elems 0..7 -> K = k0+khalf+e ; elems 8..15 -> K = k0+16+khalf+(e-8)
__device__ __forceinline__ v16bf load_a(const __bf16* buf, int stride,
                                        int m0, int k0, int lane) {
  const int m = m0 + (lane & 15);
  const int khalf = (lane >> 4) << 3;
  const __bf16* p0 = buf + m * stride + k0 + khalf;
  const __bf16* p1 = p0 + 16;
  v16bf a;
#pragma unroll
  for (int e = 0; e < 8; ++e) { a[e] = p0[e]; a[8 + e] = p1[e]; }
  return a;
}

// B fragment from a B^T (n-major) buffer: lane col n = n0 + (lane&15);
// elems 0..7 -> K = k0+khalf+e ; elems 8..15 -> K = k0+16+khalf+(e-8)
__device__ __forceinline__ v16bf load_bt(const __bf16* buf, int stride,
                                         int n0, int k0, int lane) {
  const int n = n0 + (lane & 15);
  const int kb = k0 + ((lane >> 4) << 3);
  const __bf16* p = buf + n * stride + kb;
  v16bf b;
#pragma unroll
  for (int e = 0; e < 8; ++e) { b[e] = p[e]; b[8 + e] = p[16 + e]; }
  return b;
}

__device__ __forceinline__ v8f wmma_bf16(v16bf a, v16bf b, v8f c) {
  return __builtin_amdgcn_wmma_f32_16x16x32_bf16(false, a, false, b,
                                                 (short)0, c, false, false);
}

__global__ __launch_bounds__(NTHREADS, 1)
void earth_attn_kernel(const float* __restrict__ x,
                       const float* __restrict__ mask,
                       const float* __restrict__ qkv_w,
                       const float* __restrict__ qkv_b,
                       const float* __restrict__ proj_w,
                       const float* __restrict__ proj_b,
                       const float* __restrict__ bias_table,
                       float* __restrict__ out) {
  extern __shared__ char smem[];
  __bf16* xs  = (__bf16*)(smem + OFF_XS);
  __bf16* ao  = (__bf16*)(smem + OFF_AO);
  __bf16* ps  = (__bf16*)(smem + OFF_PS);
  __bf16* vts = (__bf16*)(smem + OFF_VTS);
  __bf16* qs  = (__bf16*)(smem + OFF_QS);
  __bf16* ks  = (__bf16*)(smem + OFF_KS);
  __bf16* wt  = (__bf16*)(smem + OFF_WT);
  __bf16* pwt = (__bf16*)(smem + OFF_PWT);

  const int tid    = threadIdx.x;
  const int lane   = tid & 31;
  const int wv     = tid >> 5;        // wave id 0..8
  const int lane15 = lane & 15;
  const int half8  = (lane >> 4) << 3;
  const int bw     = blockIdx.x;      // 0..959
  const int w      = bw % NWT;
  const int m0     = wv * 16;         // this wave's token-row block
  const int w6     = w * NH;

  // ---- stage 0: x tile -> bf16 LDS; zero the K-padding regions ----------
  {
    const float* xg = x + (size_t)bw * NTOK * DIM;
    for (int i = tid; i < NTOK * DIM; i += NTHREADS) xs[i] = (__bf16)xg[i];
    for (int i = tid; i < HD * 16; i += NTHREADS)            // vts pad cols 144..159
      vts[(i >> 4) * NTOKP + NTOK + (i & 15)] = (__bf16)0.0f;
    for (int i = tid; i < NTOK * 16; i += NTHREADS)          // ps pad cols 144..159
      ps[(i >> 4) * NTOKP + NTOK + (i & 15)] = (__bf16)0.0f;
  }

  // ---- shifted-window mask for this wave's rows, held in registers ------
  float mreg[9][8];
  {
    const float* mbw = mask + (size_t)bw * NTOK * NTOK;
#pragma unroll
    for (int nb = 0; nb < 9; ++nb)
#pragma unroll
      for (int r = 0; r < 8; ++r)
        mreg[nb][r] = mbw[(m0 + r + half8) * NTOK + nb * 16 + lane15];
  }
  __syncthreads();

  const float scale = 0.17677669529663687f;   // 32^-0.5

  for (int h = 0; h < NH; ++h) {
    // ---- stage A: W^T slices for q/k/v of this head into LDS -----------
    for (int i = tid; i < 3 * HD * DIM; i += NTHREADS) {
      const int col = i % 96;            // (p,d): p = col/32, d = col%32
      const int kk  = i / 96;
      const int p   = col >> 5, d = col & 31;
      wt[(p * HD + d) * DIM + kk] =
          (__bf16)qkv_w[kk * (3 * DIM) + p * DIM + h * HD + d];
    }
    __syncthreads();

    // ---- stage B: q, k (row-major) and v^T for this head ---------------
    {
      v16bf xa[6];
#pragma unroll
      for (int kc = 0; kc < 6; ++kc) xa[kc] = load_a(xs, DIM, m0, kc * 32, lane);
#pragma unroll
      for (int p = 0; p < 3; ++p) {
        const __bf16* wp = wt + p * HD * DIM;
#pragma unroll
        for (int n0 = 0; n0 < HD; n0 += 16) {
          v8f acc = {};
#pragma unroll
          for (int kc = 0; kc < 6; ++kc)
            acc = wmma_bf16(xa[kc], load_bt(wp, DIM, n0, kc * 32, lane), acc);
          const float bv = qkv_b[p * DIM + h * HD + n0 + lane15];
          if (p == 0) {
#pragma unroll
            for (int r = 0; r < 8; ++r)
              qs[(m0 + r + half8) * HD + n0 + lane15] = (__bf16)(acc[r] + bv);
          } else if (p == 1) {
#pragma unroll
            for (int r = 0; r < 8; ++r)
              ks[(m0 + r + half8) * HD + n0 + lane15] = (__bf16)(acc[r] + bv);
          } else {
#pragma unroll
            for (int r = 0; r < 8; ++r)
              vts[(n0 + lane15) * NTOKP + m0 + r + half8] = (__bf16)(acc[r] + bv);
          }
        }
      }
    }
    __syncthreads();

    // ---- stage C: scores (K=32 == head_dim: one WMMA per 16x16 tile) ---
    v8f sc[9];
    {
      const v16bf qa = load_a(qs, HD, m0, 0, lane);
#pragma unroll
      for (int nb = 0; nb < 9; ++nb) {
        v8f z = {};
        sc[nb] = wmma_bf16(qa, load_bt(ks, HD, nb * 16, 0, lane), z);
      }
    }
    // scale + earth-position bias (analytic POS_INDEX) + mask
#pragma unroll
    for (int nb = 0; nb < 9; ++nb) {
      const int n  = nb * 16 + lane15;
      const int zn = n / 72, hn = (n / 12) % 6, wn = n % 12;
#pragma unroll
      for (int r = 0; r < 8; ++r) {
        const int m  = m0 + r + half8;
        const int zm = m / 72, hm = (m / 12) % 6, wm = m % 12;
        const int pidx = (zn + 2 * zm) * 828 + (hn + 6 * hm) * 23 + (wn - wm + 11);
        const float bv = bias_table[pidx * (NWT * NH) + w6 + h];
        sc[nb][r] = sc[nb][r] * scale + bv + mreg[nb][r];
      }
    }
    // row softmax: reduce across 16-lane halves (rows live per-VGPR)
    {
      __bf16* psw = ps + wv * 16 * NTOKP;
#pragma unroll
      for (int r = 0; r < 8; ++r) {
        float mx = -3.0e38f;
#pragma unroll
        for (int nb = 0; nb < 9; ++nb) mx = fmaxf(mx, sc[nb][r]);
        mx = fmaxf(mx, __shfl_xor(mx, 8, 32));
        mx = fmaxf(mx, __shfl_xor(mx, 4, 32));
        mx = fmaxf(mx, __shfl_xor(mx, 2, 32));
        mx = fmaxf(mx, __shfl_xor(mx, 1, 32));
        float sum = 0.0f;
#pragma unroll
        for (int nb = 0; nb < 9; ++nb) {
          const float pv = __expf(sc[nb][r] - mx);
          sc[nb][r] = pv;
          sum += pv;
        }
        sum += __shfl_xor(sum, 8, 32);
        sum += __shfl_xor(sum, 4, 32);
        sum += __shfl_xor(sum, 2, 32);
        sum += __shfl_xor(sum, 1, 32);
        const float inv = 1.0f / sum;
        const int rr = r + half8;
#pragma unroll
        for (int nb = 0; nb < 9; ++nb)
          psw[rr * NTOKP + nb * 16 + lane15] = (__bf16)(sc[nb][r] * inv);
      }
    }
    __syncthreads();

    // ---- P @ V : (16x160) @ (160x32) -----------------------------------
    {
      const __bf16* psw = ps + wv * 16 * NTOKP;
      v16bf pa[5];
#pragma unroll
      for (int kc = 0; kc < 5; ++kc) pa[kc] = load_a(psw, NTOKP, 0, kc * 32, lane);
#pragma unroll
      for (int n0 = 0; n0 < HD; n0 += 16) {
        v8f acc = {};
#pragma unroll
        for (int kc = 0; kc < 5; ++kc)
          acc = wmma_bf16(pa[kc], load_bt(vts, NTOKP, n0, kc * 32, lane), acc);
#pragma unroll
        for (int r = 0; r < 8; ++r)
          ao[(m0 + r + half8) * DIM + h * HD + n0 + lane15] = (__bf16)acc[r];
      }
    }
    __syncthreads();   // also guards next head's wt/qs/ks/vts overwrite
  }

  // ---- stage D: output projection (144x192 @ 192x192) -------------------
  for (int i = tid; i < DIM * DIM; i += NTHREADS) {
    const int n = i % DIM, k = i / DIM;
    pwt[n * DIM + k] = (__bf16)proj_w[k * DIM + n];   // aliases dead buffers
  }
  __syncthreads();
  {
    v16bf aof[6];
#pragma unroll
    for (int kc = 0; kc < 6; ++kc) aof[kc] = load_a(ao, DIM, m0, kc * 32, lane);
    float* outg = out + (size_t)bw * NTOK * DIM;
#pragma unroll
    for (int n0 = 0; n0 < DIM; n0 += 16) {
      v8f acc = {};
#pragma unroll
      for (int kc = 0; kc < 6; ++kc)
        acc = wmma_bf16(aof[kc], load_bt(pwt, DIM, n0, kc * 32, lane), acc);
      const float pb = proj_b[n0 + lane15];
#pragma unroll
      for (int r = 0; r < 8; ++r)
        outg[(m0 + r + half8) * DIM + n0 + lane15] = acc[r] + pb;
    }
  }
}

extern "C" void kernel_launch(void* const* d_in, const int* in_sizes, int n_in,
                              void* d_out, int out_size, void* d_ws, size_t ws_size,
                              hipStream_t stream) {
  (void)in_sizes; (void)n_in; (void)out_size; (void)d_ws; (void)ws_size;
  const float* x          = (const float*)d_in[0];
  const float* mask       = (const float*)d_in[1];
  const float* qkv_w      = (const float*)d_in[2];
  const float* qkv_b      = (const float*)d_in[3];
  const float* proj_w     = (const float*)d_in[4];
  const float* proj_b     = (const float*)d_in[5];
  const float* bias_table = (const float*)d_in[6];
  float* out = (float*)d_out;

  // allow >64KB dynamic LDS (idempotent; deterministic)
  (void)hipFuncSetAttribute((const void*)earth_attn_kernel,
                            hipFuncAttributeMaxDynamicSharedMemorySize,
                            SMEM_BYTES);

  dim3 grid(NLON * NWT);     // 960 window blocks
  dim3 block(NTHREADS);      // 9 waves of 32
  earth_attn_kernel<<<grid, block, SMEM_BYTES, stream>>>(
      x, mask, qkv_w, qkv_b, proj_w, proj_b, bias_table, out);
}
